// MLP_75857712382527
// MI455X (gfx1250) — matrix operand order repaired and checked by reading, not compiled, not run on hardware
//
#include <hip/hip_runtime.h>
#include <hip/hip_bf16.h>
#include <stdint.h>

typedef _Float16 v16h __attribute__((ext_vector_type(16)));
typedef _Float16 v8h  __attribute__((ext_vector_type(8)));
typedef float    v8f  __attribute__((ext_vector_type(8)));
typedef unsigned int u32x4 __attribute__((ext_vector_type(4)));
typedef int          i32x4 __attribute__((ext_vector_type(4)));
typedef int          i32x8 __attribute__((ext_vector_type(8)));

#define IN_DIM   63
#define WIDTH    256
#define ROWS_PB  128     // rows (points) per block
#define THREADS  256     // 8 wave32 waves
#define S0       328     // halfs stride, activation buffer 0 (320 used + 8 pad)
#define S1       264     // halfs stride, activation buffer 1 (256 used + 8 pad)

// LDS layout (bytes):
//   buf0 : 128 x S0 halfs = 83968   [cols 0..63 = padded input, 64..319 = x]
//   buf1 : 128 x S1 halfs = 67584
//   wbuf : 256 x 328 halfs = 167936 [transposed f16 weights, per-layer stride]
// total 319488 <= 320KB WGP LDS
#define OFF_BUF1   (ROWS_PB * S0 * 2)
#define OFF_WBUF   (OFF_BUF1 + ROWS_PB * S1 * 2)
#define SMEM_BYTES (OFF_WBUF + 256 * 328 * 2)

// Per-layer packed-weight geometry (halfs). wstride = kpad + 8 (bank-pad).
// Image in d_ws is [256][wstride] f16, identical to the LDS image -> the
// per-block stage is a single linear 2D TDM tile copy.
#define KPAD0 64
#define KPAD4 320
#define KPADM 256
constexpr int WSTR[8] = {72, 264, 264, 264, 328, 264, 264, 264};
constexpr int WOFF[8] = {0, 18432, 86016, 153600, 221184, 305152, 372736, 440320};
#define WPACK_HALFS 507904
#define WPACK_BYTES (WPACK_HALFS * 2)

#if defined(__HIP_DEVICE_COMPILE__) && __has_builtin(__builtin_amdgcn_tensor_load_to_lds) && __has_builtin(__builtin_amdgcn_s_wait_tensorcnt)
#define TDM_AVAILABLE 1
#else
#define TDM_AVAILABLE 0
#endif

// ---------------------------------------------------------------------------
// Prepass: pack W[i] (f32 row-major [fanin][256]) into d_ws as f16
// transposed Wt[n][k] with per-layer stride, zero pads, and the layer-4
// skip remap (k<63 -> W4[k], k==63 -> 0, k>=64 -> W4[k-1]) so the concat
// [in(63 pad 64), x(256)] never has to be materialized.
// ---------------------------------------------------------------------------
__global__ __launch_bounds__(THREADS)
void pack_weights_kernel(const float* __restrict__ W0, const float* __restrict__ W1,
                         const float* __restrict__ W2, const float* __restrict__ W3,
                         const float* __restrict__ W4, const float* __restrict__ W5,
                         const float* __restrict__ W6, const float* __restrict__ W7,
                         _Float16* __restrict__ wpack) {
  const float* Ws[8] = {W0, W1, W2, W3, W4, W5, W6, W7};
  constexpr int FANIN[8] = {63, 256, 256, 256, 319, 256, 256, 256};
  constexpr int KPADS[8] = {64, 256, 256, 256, 320, 256, 256, 256};
  const int layer = blockIdx.y;
  const int ws    = WSTR[layer];
  const int kpad  = KPADS[layer];
  const int fanin = FANIN[layer];
  const int total = 256 * ws;
  for (int idx = blockIdx.x * THREADS + threadIdx.x; idx < total;
       idx += gridDim.x * THREADS) {
    const int n = idx / ws;
    const int k = idx - n * ws;
    float v = 0.0f;
    if (k < kpad) {
      int src;
      if (layer == 4) src = (k < 63) ? k : ((k == 63) ? -1 : k - 1);
      else            src = (k < fanin) ? k : -1;
      if (src >= 0) v = Ws[layer][(size_t)src * 256 + n];
    }
    wpack[(size_t)WOFF[layer] + idx] = (_Float16)v;
  }
}

// ---------------------------------------------------------------------------
// TDM: single DMA of one packed layer image (256 * wstride halfs) from
// global into LDS at byte offset lds_off. 2D tile, 8-byte elements.
// Issued by one wave (EXEC ignored by TDM; avoid 8x redundant issues).
// ---------------------------------------------------------------------------
__device__ __forceinline__ void tdm_copy_w(const _Float16* __restrict__ gsrc,
                                           unsigned lds_off, int wstride) {
#if TDM_AVAILABLE
  if (threadIdx.x < 32) {
    const unsigned long long ga = (unsigned long long)(uintptr_t)gsrc;
    const unsigned L = (unsigned)(wstride >> 2);   // row length in 8B units
    u32x4 g0;
    g0[0] = 1u;                                    // count=1, user descriptor
    g0[1] = lds_off;                               // lds_addr (bytes)
    g0[2] = (unsigned)ga;                          // global_addr[31:0]
    g0[3] = (unsigned)(ga >> 32) | (2u << 30);     // global_addr[56:32] | type=2
    i32x8 g1;
    g1[0] = (int)(3u << 16);                       // wg_mask=0, data_size=3 (8B)
    g1[1] = (int)(L << 16);                        // tensor_dim0[15:0]
    g1[2] = (int)(256u << 16);                     // tensor_dim0 hi=0 | tensor_dim1=256
    g1[3] = (int)(L << 16);                        // tensor_dim1 hi=0 | tile_dim0=L
    g1[4] = 256;                                   // tile_dim1=256 | tile_dim2=0
    g1[5] = (int)L;                                // tensor_dim0_stride lo32
    g1[6] = (int)((L * 256u) << 16);               // stride hi=0 | dim1_stride lo
    g1[7] = 0;                                     // dim1_stride hi
    i32x4 gz = {0, 0, 0, 0};                       // groups 2/3: 2D tensor
#if __clang_major__ >= 23
    i32x8 gz8 = {0, 0, 0, 0, 0, 0, 0, 0};
    __builtin_amdgcn_tensor_load_to_lds(g0, g1, gz, gz, gz8, 0);
#else
    __builtin_amdgcn_tensor_load_to_lds(g0, g1, gz, gz, 0);
#endif
    __builtin_amdgcn_s_wait_tensorcnt(0);
  }
#else
  (void)gsrc; (void)lds_off; (void)wstride;
#endif
}

// Fallback stage (also used when ws_size is too small): f32 global ->
// f16 transposed LDS with optional layer-4 remap.
__device__ __forceinline__ void load_weights(const float* __restrict__ W,
                                             int fanin, int kpad, int wstride,
                                             _Float16* wlds, bool insert63) {
  const int total = kpad * 256;
  for (int idx = threadIdx.x; idx < total; idx += THREADS) {
    const int k = idx >> 8;
    const int n = idx & 255;
    int src;
    if (insert63) src = (k < 63) ? k : ((k == 63) ? -1 : k - 1);
    else          src = (k < fanin) ? k : -1;
    const float v = (src >= 0) ? W[(size_t)src * 256 + n] : 0.0f;
    wlds[(size_t)n * wstride + k] = (_Float16)v;
  }
}

// Warm L2 with the next layer's weight source while this layer computes.
__device__ __forceinline__ void prefetch_bytes(const void* __restrict__ p, int bytes) {
  const int lines = bytes >> 7;
  for (int l = threadIdx.x; l < lines; l += THREADS)
    __builtin_prefetch((const char*)p + l * 128, 0, 1);   // global_prefetch_b8
}

// ---------------------------------------------------------------------------
// One dense layer on this block's 128 rows.
//  A: alds row-major f16 [128][astride], K = KPAD (zero padded)
//  B: wlds f16 transposed [256][wstride]
//  out: ReLU->f16 LDS (gout==nullptr) or bias-only f32 global (last layer).
// ---------------------------------------------------------------------------
template <int KPAD>
__device__ __forceinline__ void layer_compute(
    const _Float16* __restrict__ alds, int astride,
    const _Float16* __restrict__ wlds, int wstride,
    const float* __restrict__ bias,
    _Float16* outlds, int ostride,
    float* __restrict__ gout, size_t row0, int Ntot) {
  constexpr int KB = KPAD / 32;
  const int lane = threadIdx.x & 31;
  const int wv   = threadIdx.x >> 5;   // 0..7 -> 16-row strip
  const int l15  = lane & 15;
  const int hi8  = (lane >> 4) * 8;    // 0 | 8
  const int mrow = wv * 16;

  // Preload A fragments (ISA 16-bit A 16x32 layout):
  // lanes 0-15: K pairs {0..7},{16..23}; lanes 16-31: {8..15},{24..31}.
  v16h a[KB];
  const _Float16* ap = alds + (size_t)(mrow + l15) * astride + hi8;
#pragma unroll
  for (int kb = 0; kb < KB; ++kb) {
    union { v16h v; v8h h[2]; } u;
    u.h[0] = *(const v8h*)(ap + kb * 32);        // ds_load_b128
    u.h[1] = *(const v8h*)(ap + kb * 32 + 16);   // ds_load_b128
    a[kb] = u.v;
  }

  for (int nt = 0; nt < 16; ++nt) {
    const int col = nt * 16 + l15;
    // B layout: lane<16 -> col=lane, K {0..7},{16..23}; lane>=16 -> K {8..15},{24..31}.
    const _Float16* wp = wlds + (size_t)col * wstride + hi8;
    v8f c = {};
#pragma unroll
    for (int kb = 0; kb < KB; ++kb) {
      union { v16h v; v8h h[2]; } b;
      b.h[0] = *(const v8h*)(wp + kb * 32);
      b.h[1] = *(const v8h*)(wp + kb * 32 + 16);
      c = __builtin_amdgcn_wmma_f32_16x16x32_f16(
          /*neg_a=*/false, a[kb], /*neg_b=*/false, b.v,
          /*c_mod=*/(short)0, c, /*reuse_a=*/false, /*reuse_b=*/false);
    }
    const float bv = bias[col];
    if (gout == nullptr) {
      // C/D layout: VGPR r -> row (mrow + hi8 + r), col = nt*16 + (lane&15)
      _Float16* op = outlds + (size_t)(mrow + hi8) * ostride + col;
#pragma unroll
      for (int r = 0; r < 8; ++r) {
        float v = c[r] + bv;
        v = v > 0.0f ? v : 0.0f;                 // ReLU
        op[(size_t)r * ostride] = (_Float16)v;
      }
    } else {
#pragma unroll
      for (int r = 0; r < 8; ++r) {
        const size_t row = row0 + (size_t)(mrow + hi8 + r);
        if (row < (size_t)Ntot) gout[row * 256 + col] = c[r] + bv;
      }
    }
  }
}

__global__ __launch_bounds__(THREADS, 1)
void mlp_fused_kernel(const float* __restrict__ in,
                      const float* __restrict__ W0, const float* __restrict__ B0,
                      const float* __restrict__ W1, const float* __restrict__ B1,
                      const float* __restrict__ W2, const float* __restrict__ B2,
                      const float* __restrict__ W3, const float* __restrict__ B3,
                      const float* __restrict__ W4, const float* __restrict__ B4,
                      const float* __restrict__ W5, const float* __restrict__ B5,
                      const float* __restrict__ W6, const float* __restrict__ B6,
                      const float* __restrict__ W7, const float* __restrict__ B7,
                      const _Float16* __restrict__ wpack,
                      float* __restrict__ out, int N) {
  extern __shared__ char smem[];
  _Float16* buf0 = (_Float16*)smem;
  _Float16* buf1 = (_Float16*)(smem + OFF_BUF1);
  _Float16* wbuf = (_Float16*)(smem + OFF_WBUF);
  const size_t row0 = (size_t)blockIdx.x * ROWS_PB;
  const bool use_tdm = TDM_AVAILABLE && (wpack != nullptr);

  // Stage input: 128 x 64 f16 (col 63 zero-padded) into buf0[:, 0:64].
  for (int idx = threadIdx.x; idx < ROWS_PB * 64; idx += THREADS) {
    const int r = idx >> 6, cidx = idx & 63;
    const size_t grow = row0 + r;
    float v = 0.0f;
    if (cidx < IN_DIM && grow < (size_t)N) v = in[grow * IN_DIM + cidx];
    buf0[(size_t)r * S0 + cidx] = (_Float16)v;
  }
  // Layer 0 weights overlap with input staging.
  if (use_tdm) tdm_copy_w(wpack + WOFF[0], OFF_WBUF, WSTR[0]);
  else         load_weights(W0, 63, KPAD0, WSTR[0], wbuf, false);
  __syncthreads();

  if (use_tdm) prefetch_bytes(wpack + WOFF[1], 256 * WSTR[1] * 2);
  else         prefetch_bytes(W1, 256 * 256 * 4);
  layer_compute<KPAD0>(buf0, S0, wbuf, WSTR[0], B0, buf1, S1, nullptr, 0, 0);
  __syncthreads();

  if (use_tdm) tdm_copy_w(wpack + WOFF[1], OFF_WBUF, WSTR[1]);
  else         load_weights(W1, 256, KPADM, WSTR[1], wbuf, false);
  __syncthreads();
  if (use_tdm) prefetch_bytes(wpack + WOFF[2], 256 * WSTR[2] * 2);
  else         prefetch_bytes(W2, 256 * 256 * 4);
  layer_compute<KPADM>(buf1, S1, wbuf, WSTR[1], B1, buf0 + 64, S0, nullptr, 0, 0);
  __syncthreads();

  if (use_tdm) tdm_copy_w(wpack + WOFF[2], OFF_WBUF, WSTR[2]);
  else         load_weights(W2, 256, KPADM, WSTR[2], wbuf, false);
  __syncthreads();
  if (use_tdm) prefetch_bytes(wpack + WOFF[3], 256 * WSTR[3] * 2);
  else         prefetch_bytes(W3, 256 * 256 * 4);
  layer_compute<KPADM>(buf0 + 64, S0, wbuf, WSTR[2], B2, buf1, S1, nullptr, 0, 0);
  __syncthreads();

  if (use_tdm) tdm_copy_w(wpack + WOFF[3], OFF_WBUF, WSTR[3]);
  else         load_weights(W3, 256, KPADM, WSTR[3], wbuf, false);
  __syncthreads();
  if (use_tdm) prefetch_bytes(wpack + WOFF[4], 256 * WSTR[4] * 2);
  else         prefetch_bytes(W4, 319 * 256 * 4);
  layer_compute<KPADM>(buf1, S1, wbuf, WSTR[3], B3, buf0 + 64, S0, nullptr, 0, 0);
  __syncthreads();

  // Skip layer: A = buf0 full 320 wide = [input(64 padded) | x(256)].
  if (use_tdm) tdm_copy_w(wpack + WOFF[4], OFF_WBUF, WSTR[4]);
  else         load_weights(W4, 319, KPAD4, WSTR[4], wbuf, true);
  __syncthreads();
  if (use_tdm) prefetch_bytes(wpack + WOFF[5], 256 * WSTR[5] * 2);
  else         prefetch_bytes(W5, 256 * 256 * 4);
  layer_compute<KPAD4>(buf0, S0, wbuf, WSTR[4], B4, buf1, S1, nullptr, 0, 0);
  __syncthreads();

  if (use_tdm) tdm_copy_w(wpack + WOFF[5], OFF_WBUF, WSTR[5]);
  else         load_weights(W5, 256, KPADM, WSTR[5], wbuf, false);
  __syncthreads();
  if (use_tdm) prefetch_bytes(wpack + WOFF[6], 256 * WSTR[6] * 2);
  else         prefetch_bytes(W6, 256 * 256 * 4);
  layer_compute<KPADM>(buf1, S1, wbuf, WSTR[5], B5, buf0 + 64, S0, nullptr, 0, 0);
  __syncthreads();

  if (use_tdm) tdm_copy_w(wpack + WOFF[6], OFF_WBUF, WSTR[6]);
  else         load_weights(W6, 256, KPADM, WSTR[6], wbuf, false);
  __syncthreads();
  if (use_tdm) prefetch_bytes(wpack + WOFF[7], 256 * WSTR[7] * 2);
  else         prefetch_bytes(W7, 256 * 256 * 4);
  layer_compute<KPADM>(buf0 + 64, S0, wbuf, WSTR[6], B6, buf1, S1, nullptr, 0, 0);
  __syncthreads();

  if (use_tdm) tdm_copy_w(wpack + WOFF[7], OFF_WBUF, WSTR[7]);
  else         load_weights(W7, 256, KPADM, WSTR[7], wbuf, false);
  __syncthreads();
  layer_compute<KPADM>(buf1, S1, wbuf, WSTR[7], B7, nullptr, 0, out, row0, N);
}

extern "C" void kernel_launch(void* const* d_in, const int* in_sizes, int n_in,
                              void* d_out, int out_size, void* d_ws, size_t ws_size,
                              hipStream_t stream) {
  (void)n_in; (void)out_size;
  const float* in = (const float*)d_in[0];
  const float* W[8];
  const float* B[8];
  for (int i = 0; i < 8; ++i) {
    W[i] = (const float*)d_in[1 + 2 * i];
    B[i] = (const float*)d_in[2 + 2 * i];
  }
  float* out = (float*)d_out;
  const int N = in_sizes[0] / IN_DIM;          // 524288
  const int grid = (N + ROWS_PB - 1) / ROWS_PB;

  _Float16* wpack = nullptr;
  if (d_ws != nullptr && ws_size >= (size_t)WPACK_BYTES) {
    wpack = (_Float16*)d_ws;
    pack_weights_kernel<<<dim3(328, 8), THREADS, 0, stream>>>(
        W[0], W[1], W[2], W[3], W[4], W[5], W[6], W[7], wpack);
  }

  (void)hipFuncSetAttribute((const void*)mlp_fused_kernel,
                            hipFuncAttributeMaxDynamicSharedMemorySize,
                            SMEM_BYTES);
  mlp_fused_kernel<<<grid, THREADS, SMEM_BYTES, stream>>>(
      in, W[0], B[0], W[1], B[1], W[2], B[2], W[3], B[3],
      W[4], B[4], W[5], B[5], W[6], B[6], W[7], B[7], wpack, out, N);
}
